// SequencePredictor_54288386621510
// MI455X (gfx1250) — compile-verified, matrix operand order checked by reading, not run on hardware
//
#include <hip/hip_runtime.h>
#include <hip/hip_bf16.h>
#include <stdint.h>

// ---------------------------------------------------------------------------
// Problem constants (from reference): B=32, TX=TY=64, D=H=512, V=32000
// ---------------------------------------------------------------------------
#define BATCH 32
#define TSEQ 64
#define DIM 512
#define HID 512
#define NGATE 2048   // 4*H
#define KW 1024      // D+H (rows of W)
#define VOCAB 32000
#define NWG_LSTM 32  // persistent workgroups for recurrence (16 hidden cols each)

typedef __attribute__((ext_vector_type(16))) __bf16 v16bf;
typedef __attribute__((ext_vector_type(8)))  float  v8f;

union ABFrag { v16bf v; unsigned u[8]; };
union CFrag  { v8f v; float f[8]; };

__device__ inline unsigned short f2bf_bits(float f) {
  union { float f; unsigned u; } x; x.f = f;
  unsigned r = x.u + 0x7FFFu + ((x.u >> 16) & 1u);
  return (unsigned short)(r >> 16);
}
__device__ inline float sigf(float x) { return 1.0f / (1.0f + __expf(-x)); }

// K-dimension element mapping within a 16-bit WMMA A/B fragment.
// Fragment element e (0..15): K(e) = (e<8 ? e : e+8) + (lane/16)*8.
// dword d loads elements (2d, 2d+1): contiguous K pair at offset koff(d).
__device__ inline int koff(int d) { return (d < 4) ? 2 * d : 2 * d + 8; }

// ---------------------------------------------------------------------------
// f32 [K,N] -> bf16 [N,K] transpose (LDS tiled). Makes every WMMA B-operand
// K-contiguous in memory so fragments load as packed dwords.
// ---------------------------------------------------------------------------
__global__ void k_transpose_bf16(const float* __restrict__ in,
                                 unsigned short* __restrict__ out,
                                 int K, int N) {
  __shared__ float tile[32][33];
  const int n0 = blockIdx.x * 32, k0 = blockIdx.y * 32;
  const int tx = threadIdx.x, ty = threadIdx.y;  // 32 x 8
#pragma unroll
  for (int i = 0; i < 32; i += 8)
    tile[ty + i][tx] = in[(size_t)(k0 + ty + i) * N + n0 + tx];
  __syncthreads();
#pragma unroll
  for (int i = 0; i < 32; i += 8)
    out[(size_t)(n0 + ty + i) * K + k0 + tx] = f2bf_bits(tile[tx][ty + i]);
}

// ---------------------------------------------------------------------------
// Embedding gather: X[t*B+b, :] = bf16(E[ids[b,t], :])
// ---------------------------------------------------------------------------
__global__ void k_embed(const int* __restrict__ ids, const float* __restrict__ E,
                        unsigned short* __restrict__ X, int T) {
  const int s = blockIdx.x;           // s = t*B + b
  const int t = s >> 5, b = s & 31;
  const int id = ids[b * T + t];
  const float* src = E + (size_t)id * DIM;
  unsigned short* dst = X + (size_t)s * DIM;
  for (int d = threadIdx.x; d < DIM; d += blockDim.x) dst[d] = f2bf_bits(src[d]);
}

// ---------------------------------------------------------------------------
// bf16 GEMM: C[M,N](f32) = A[M,K](bf16 rowmajor) x Bt[N,K](bf16, K-contig).
// 256 threads = 8 waves, wave tile 32x32 (2x2 WMMA frags), WG tile 64x128.
// remap!=0: output row s=t*B+b is scattered to row (b*TY + t)  (logits [B,TY,V]).
// ---------------------------------------------------------------------------
__global__ void k_gemm(const unsigned short* __restrict__ A, int lda,
                       const unsigned short* __restrict__ Bt, int ldb, int kstart,
                       float* __restrict__ C, int ldc,
                       int M, int N, int K, int remap) {
  const int w = threadIdx.x >> 5, lane = threadIdx.x & 31;
  const int half = lane >> 4, r = lane & 15;
  const int row0 = blockIdx.y * 64 + (w & 1) * 32;
  const int col0 = blockIdx.x * 128 + (w >> 1) * 32;

  CFrag acc[2][2];
#pragma unroll
  for (int i = 0; i < 2; ++i)
#pragma unroll
    for (int j = 0; j < 2; ++j)
#pragma unroll
      for (int e = 0; e < 8; ++e) acc[i][j].f[e] = 0.0f;

  for (int kk = 0; kk < K; kk += 32) {
    ABFrag a[2], b[2];
    const unsigned short* ap0 = A + (size_t)(row0 + r) * lda + kk + half * 8;
    const unsigned short* ap1 = ap0 + (size_t)16 * lda;
    const unsigned short* bp0 = Bt + (size_t)(col0 + r) * ldb + kstart + kk + half * 8;
    const unsigned short* bp1 = bp0 + (size_t)16 * ldb;
#pragma unroll
    for (int d = 0; d < 8; ++d) {
      const int ko = koff(d);
      a[0].u[d] = *(const unsigned*)(ap0 + ko);
      a[1].u[d] = *(const unsigned*)(ap1 + ko);
      b[0].u[d] = *(const unsigned*)(bp0 + ko);
      b[1].u[d] = *(const unsigned*)(bp1 + ko);
    }
    // CDNA5 speculative prefetch of the next K-chunk of the streamed operand
    __builtin_prefetch(bp0 + 32, 0, 1);
    __builtin_prefetch(bp1 + 32, 0, 1);
    acc[0][0].v = __builtin_amdgcn_wmma_f32_16x16x32_bf16(false, a[0].v, false, b[0].v,
                                                          (short)0, acc[0][0].v, false, false);
    acc[0][1].v = __builtin_amdgcn_wmma_f32_16x16x32_bf16(false, a[0].v, false, b[1].v,
                                                          (short)0, acc[0][1].v, false, false);
    acc[1][0].v = __builtin_amdgcn_wmma_f32_16x16x32_bf16(false, a[1].v, false, b[0].v,
                                                          (short)0, acc[1][0].v, false, false);
    acc[1][1].v = __builtin_amdgcn_wmma_f32_16x16x32_bf16(false, a[1].v, false, b[1].v,
                                                          (short)0, acc[1][1].v, false, false);
  }

#pragma unroll
  for (int i = 0; i < 2; ++i)
#pragma unroll
    for (int j = 0; j < 2; ++j)
#pragma unroll
      for (int e = 0; e < 8; ++e) {
        const int m = row0 + i * 16 + half * 8 + e;
        const int n = col0 + j * 16 + r;
        size_t idx;
        if (remap) {
          // s = t*BATCH + b  ->  logits row b*TSEQ + t
          idx = ((size_t)((m & 31) * TSEQ + (m >> 5))) * (size_t)ldc + n;
        } else {
          idx = (size_t)m * (size_t)ldc + n;
        }
        C[idx] = acc[i][j].f[e];
      }
}

// ---------------------------------------------------------------------------
// Persistent LSTM recurrence. 32 WGs x 256 thr; WG owns 16 hidden columns
// (its i/j/f/o gate groups). Per step: z = Zx[t] + h_bf16 @ W_h (WMMA),
// gate nonlinearity, masked state update, grid barrier. h is ping-pong
// double-buffered in bf16 so step t+1 reads never race step t writes.
// ---------------------------------------------------------------------------
__global__ void k_lstm(const float* __restrict__ Zx,            // [T*B, 2048]
                       const unsigned short* __restrict__ Wt,   // [2048, 1024]
                       const int* __restrict__ len,             // [B]
                       const float* __restrict__ bias,          // [2048]
                       float* __restrict__ c, float* __restrict__ h,
                       unsigned short* __restrict__ hbf,        // [2][B*H]
                       unsigned short* __restrict__ hs,         // [T*B, H] or null
                       int T, unsigned* __restrict__ barcnt, int nwg) {
  __shared__ float zbuf[8][16][16];
  const int w = threadIdx.x >> 5, lane = threadIdx.x & 31;
  const int half = lane >> 4, r = lane & 15;
  const int m = w & 1;        // M-tile (batch 0..15 / 16..31)
  const int g = w >> 1;       // gate index: 0=i 1=j 2=f 3=o
  const int h0 = blockIdx.x * 16;
  const int n = g * HID + h0 + r;                       // gate column (B-frag col)
  const unsigned short* bbase = Wt + (size_t)n * KW + DIM + half * 8;  // recurrent rows

  for (int t = 0; t < T; ++t) {
    const unsigned short* hread = hbf + (size_t)(t & 1) * (BATCH * HID);
    unsigned short* hwrite = hbf + (size_t)((t + 1) & 1) * (BATCH * HID);

    CFrag acc;
#pragma unroll
    for (int e = 0; e < 8; ++e) acc.f[e] = 0.0f;

    for (int kk = 0; kk < HID; kk += 32) {
      ABFrag a, b;
      const unsigned short* ap = hread + (size_t)(m * 16 + r) * HID + kk + half * 8;
      const unsigned short* bp = bbase + kk;
#pragma unroll
      for (int d = 0; d < 8; ++d) {
        const int ko = koff(d);
        a.u[d] = *(const unsigned*)(ap + ko);
        b.u[d] = *(const unsigned*)(bp + ko);
      }
      acc.v = __builtin_amdgcn_wmma_f32_16x16x32_bf16(false, a.v, false, b.v,
                                                      (short)0, acc.v, false, false);
    }

    // z = recurrent + precomputed input part + bias; stage to LDS for gate mix
#pragma unroll
    for (int e = 0; e < 8; ++e) {
      const int row = half * 8 + e;      // row within this 16-row M-tile
      const int bidx = m * 16 + row;     // batch index
      zbuf[w][row][r] = acc.f[e] + Zx[((size_t)t * BATCH + bidx) * NGATE + n] + bias[n];
    }
    __syncthreads();

    // elementwise cell update for this WG's [32 batch x 16 hidden] block
    for (int idx = threadIdx.x; idx < BATCH * 16; idx += blockDim.x) {
      const int b = idx >> 4, col = idx & 15;
      const int mm = b >> 4, rr = b & 15;
      const float zi = zbuf[0 + mm][rr][col];
      const float zj = zbuf[2 + mm][rr][col];
      const float zf = zbuf[4 + mm][rr][col];
      const float zo = zbuf[6 + mm][rr][col];
      const size_t off = (size_t)b * HID + h0 + col;
      const float co = c[off], ho = h[off];
      const float cn = co * sigf(zf + 1.0f) + sigf(zi) * tanhf(zj);
      const float hn = tanhf(cn) * sigf(zo);
      const bool mk = t < len[b];
      const float cw = mk ? cn : co;
      const float hw = mk ? hn : ho;
      c[off] = cw;
      h[off] = hw;
      hwrite[off] = f2bf_bits(hw);
      if (hs)  // emitted output: zeroed past sequence length (drives proj mask)
        hs[((size_t)t * BATCH + b) * HID + h0 + col] = mk ? f2bf_bits(hn) : (unsigned short)0;
    }

    // device-scope grid barrier (monotonic arrive count, sleep-wait)
    __threadfence();
    __syncthreads();
    if (threadIdx.x == 0) {
      atomicAdd(barcnt, 1u);
      const unsigned tgt = (unsigned)nwg * (unsigned)(t + 1);
      while (__hip_atomic_load(barcnt, __ATOMIC_ACQUIRE, __HIP_MEMORY_SCOPE_AGENT) < tgt)
        __builtin_amdgcn_s_sleep(2);
      __threadfence();
    }
    __syncthreads();
  }
}

// ---------------------------------------------------------------------------
// Host orchestration
// ---------------------------------------------------------------------------
extern "C" void kernel_launch(void* const* d_in, const int* in_sizes, int n_in,
                              void* d_out, int out_size, void* d_ws, size_t ws_size,
                              hipStream_t stream) {
  const int*   enc_ids = (const int*)d_in[0];
  const int*   dec_ids = (const int*)d_in[1];
  const int*   len_enc = (const int*)d_in[2];
  const int*   len_dec = (const int*)d_in[3];
  const float* E       = (const float*)d_in[4];
  const float* enc_W   = (const float*)d_in[5];
  const float* enc_b   = (const float*)d_in[6];
  const float* dec_W   = (const float*)d_in[7];
  const float* dec_b   = (const float*)d_in[8];
  const float* proj_W  = (const float*)d_in[9];
  float* out = (float*)d_out;
  char* ws = (char*)d_ws;

  const int SB = TSEQ * BATCH;  // 2048 sequence rows
  // workspace layout (bytes)
  size_t off = 0;
  auto take = [&](size_t bytes) { size_t o = off; off += (bytes + 255) & ~(size_t)255; return o; };
  const size_t oXenc  = take((size_t)SB * DIM * 2);
  const size_t oXdec  = take((size_t)SB * DIM * 2);
  const size_t oWtEnc = take((size_t)NGATE * KW * 2);
  const size_t oWtDec = take((size_t)NGATE * KW * 2);
  const size_t oWtPrj = take((size_t)VOCAB * HID * 2);
  const size_t oZxEnc = take((size_t)SB * NGATE * 4);
  const size_t oZxDec = take((size_t)SB * NGATE * 4);
  const size_t oC     = take((size_t)BATCH * HID * 4);
  const size_t oH     = take((size_t)BATCH * HID * 4);
  const size_t oHbf   = take((size_t)2 * BATCH * HID * 2);
  const size_t oHs    = take((size_t)SB * HID * 2);
  const size_t oBar   = take(256);
  if (ws_size < off) return;  // insufficient scratch

  unsigned short* Xenc  = (unsigned short*)(ws + oXenc);
  unsigned short* Xdec  = (unsigned short*)(ws + oXdec);
  unsigned short* WtEnc = (unsigned short*)(ws + oWtEnc);
  unsigned short* WtDec = (unsigned short*)(ws + oWtDec);
  unsigned short* WtPrj = (unsigned short*)(ws + oWtPrj);
  float* ZxEnc = (float*)(ws + oZxEnc);
  float* ZxDec = (float*)(ws + oZxDec);
  float* cSt   = (float*)(ws + oC);
  float* hSt   = (float*)(ws + oH);
  unsigned short* hbf = (unsigned short*)(ws + oHbf);
  unsigned short* hs  = (unsigned short*)(ws + oHs);
  unsigned* bar0 = (unsigned*)(ws + oBar);
  unsigned* bar1 = bar0 + 32;

  // zero initial state (c, h, both h bf16 buffers) and barrier counters
  hipMemsetAsync(ws + oC, 0, (oHbf - oC) + (size_t)2 * BATCH * HID * 2, stream);
  hipMemsetAsync(ws + oBar, 0, 256, stream);

  // weight convert+transpose -> bf16 [N,K]
  {
    dim3 blk(32, 8);
    k_transpose_bf16<<<dim3(NGATE / 32, KW / 32), blk, 0, stream>>>(enc_W, WtEnc, KW, NGATE);
    k_transpose_bf16<<<dim3(NGATE / 32, KW / 32), blk, 0, stream>>>(dec_W, WtDec, KW, NGATE);
    k_transpose_bf16<<<dim3(VOCAB / 32, HID / 32), blk, 0, stream>>>(proj_W, WtPrj, HID, VOCAB);
  }

  // embedding gathers (sequence-major [t*B+b, D], bf16)
  k_embed<<<SB, 128, 0, stream>>>(enc_ids, E, Xenc, TSEQ);
  k_embed<<<SB, 128, 0, stream>>>(dec_ids, E, Xdec, TSEQ);

  // hoisted input GEMMs: Zx = X @ W[0:D, :]   (M=2048, N=2048, K=512)
  k_gemm<<<dim3(NGATE / 128, SB / 64), 256, 0, stream>>>(
      Xenc, DIM, WtEnc, KW, /*kstart=*/0, ZxEnc, NGATE, SB, NGATE, DIM, 0);
  k_gemm<<<dim3(NGATE / 128, SB / 64), 256, 0, stream>>>(
      Xdec, DIM, WtDec, KW, /*kstart=*/0, ZxDec, NGATE, SB, NGATE, DIM, 0);

  // recurrences (persistent, grid-barriered)
  k_lstm<<<NWG_LSTM, 256, 0, stream>>>(ZxEnc, WtEnc, len_enc, enc_b,
                                       cSt, hSt, hbf, nullptr, TSEQ, bar0, NWG_LSTM);
  k_lstm<<<NWG_LSTM, 256, 0, stream>>>(ZxDec, WtDec, len_dec, dec_b,
                                       cSt, hSt, hbf, hs, TSEQ, bar1, NWG_LSTM);

  // vocab projection with [B,TY,V] scatter: M=2048, N=32000, K=512
  k_gemm<<<dim3(VOCAB / 128, SB / 64), 256, 0, stream>>>(
      hs, HID, WtPrj, HID, /*kstart=*/0, out, VOCAB, SB, VOCAB, HID, 1);

  (void)in_sizes; (void)n_in; (void)out_size;
}
